// TreeLSTM_35725537968255
// MI455X (gfx1250) — matrix operand order repaired and checked by reading, not compiled, not run on hardware
//
#include <hip/hip_runtime.h>

// ---------------------------------------------------------------------------
// TreeLSTM (shift-reduce) for MI455X / gfx1250.
// Phase 1: batch GEMM (bf16 WMMA) producing h_buf / c_buf, fused epilogue.
// Phase 2: persistent 16-WG cluster running the 511-step dependent chain,
//          synchronized with CDNA5 cluster barriers; Wr slice staged into LDS
//          by the Tensor Data Mover (descriptor-driven DMA with HW padding),
//          falling back to plain loads if the TDM builtin is unavailable.
// ---------------------------------------------------------------------------

#if defined(__has_builtin)
#  if __has_builtin(__builtin_amdgcn_tensor_load_to_lds)
#    define HAVE_TDM 1
#  endif
#endif
#ifndef HAVE_TDM
#  define HAVE_TDM 0
#endif

typedef __attribute__((ext_vector_type(8)))  float        v8f;
typedef __attribute__((ext_vector_type(16))) __bf16       v16bf;
typedef __attribute__((ext_vector_type(8)))  __bf16       bf16x8;
typedef __attribute__((ext_vector_type(4)))  unsigned int v4u;
typedef __attribute__((ext_vector_type(4)))  int          v4i;
typedef __attribute__((ext_vector_type(8)))  int          v8i;

constexpr int Bb = 128;   // batch
constexpr int Tt = 256;   // sequence length
constexpr int Ee = 256;   // embed
constexpr int Hh = 256;   // hidden
constexpr int NSLICE = 16;           // cluster workgroups (cols of H per WG = 16)
constexpr int WR_PAD = 264;          // 256 + 8 bf16 pad -> conflict-free ds_load_b128

union FragU { bf16x8 h[2]; v16bf v; };

__device__ __forceinline__ int   imax(int a, int b) { return a > b ? a : b; }
__device__ __forceinline__ float sigmf(float x)     { return 1.0f / (1.0f + __expf(-x)); }

__device__ __forceinline__ v8f wmma_bf16(v16bf a, v16bf b, v8f c) {
  // D = A(16x32 bf16) * B(32x16 bf16) + C(16x16 f32)
  return __builtin_amdgcn_wmma_f32_16x16x32_bf16(
      /*neg_a=*/false, a, /*neg_b=*/false, b,
      /*c_mod=*/(short)0, c, /*reuse_a=*/false, /*reuse_b=*/false);
}

// A fragment (16x32, bf16): lane holds row m = lane&15, half = lane>>4.
// Per ISA layout: element e -> k = (e<8 ? 8*half + e : 16 + 8*half + (e-8)).
__device__ __forceinline__ v16bf ld_a_frag(const __bf16* row, int k0, int half) {
  FragU f;
  const __bf16* p = row + k0 + 8 * half;
  f.h[0] = *(const bf16x8*)(p);
  f.h[1] = *(const bf16x8*)(p + 16);
  return f.v;
}
// B fragment (32x16, bf16): lane holds col n = lane&15 (== a row of W, since
// B = W^T), element e -> k = k0 + (e<8 ? 8*half + e : 16 + 8*half + (e-8)).
__device__ __forceinline__ v16bf ld_b_frag(const __bf16* row, int k0, int half) {
  FragU f;
  const __bf16* p = row + k0 + 8 * half;
  f.h[0] = *(const bf16x8*)(p);
  f.h[1] = *(const bf16x8*)(p + 16);
  return f.v;
}

// ---------------------------------------------------------------------------
__global__ void k_cvt_bf16(const float* __restrict__ src, __bf16* __restrict__ dst, int n) {
  int i = blockIdx.x * blockDim.x + threadIdx.x;
  if (i < n) dst[i] = (__bf16)src[i];
}

__global__ void k_zero_stack(__bf16* hs, float* cs, int n) {
  int i = blockIdx.x * blockDim.x + threadIdx.x;
  if (i < n) { hs[i] = (__bf16)0.0f; cs[i] = 0.0f; }
}

// ---------------------------------------------------------------------------
// Phase 1: for every (b,t) row m: cx = x@Wx^T + bx ; gx = x@Wg^T + bg ;
//          c_buf = cx (f32), h_buf = sigmoid(gx)*tanh(cx) (bf16).
__global__ __launch_bounds__(256) void k_buffers(
    const __bf16* __restrict__ xb, const __bf16* __restrict__ Wxb,
    const __bf16* __restrict__ Wgb, const float* __restrict__ bx,
    const float* __restrict__ bg, float* __restrict__ c_buf,
    __bf16* __restrict__ h_buf) {
  int lane = threadIdx.x & 31, wave = threadIdx.x >> 5;
  int half = lane >> 4, l16 = lane & 15;
  int mt = blockIdx.x * 8 + wave;        // 16-row tile index over B*T rows
  int n0 = blockIdx.y * 16;              // h-col tile base

  const __bf16* arow  = xb  + (size_t)(mt * 16 + l16) * Ee;
  const __bf16* wxrow = Wxb + (size_t)(n0 + l16) * Ee;
  const __bf16* wgrow = Wgb + (size_t)(n0 + l16) * Ee;

  v8f accC = {}, accG = {};
  for (int kt = 0; kt < Ee / 32; ++kt) {
    int k0 = kt * 32;
    v16bf a  = ld_a_frag(arow,  k0, half);
    v16bf bX = ld_b_frag(wxrow, k0, half);
    v16bf bG = ld_b_frag(wgrow, k0, half);
    accC = wmma_bf16(a, bX, accC);
    accG = wmma_bf16(a, bG, accG);
  }

  int col = n0 + l16;
  float bxv = bx[col], bgv = bg[col];
#pragma unroll
  for (int i = 0; i < 8; ++i) {
    int row = mt * 16 + i + 8 * half;    // C/D layout: VGPR i -> M = i + 8*half
    float cx = accC[i] + bxv;
    float gx = accG[i] + bgv;
    float hv = sigmf(gx) * tanhf(cx);
    c_buf[(size_t)row * Hh + col] = cx;
    h_buf[(size_t)row * Hh + col] = (__bf16)hv;
  }
}

// ---------------------------------------------------------------------------
// Phase 2: persistent cluster kernel. WG p owns h-cols [16p, 16p+16).
__global__ __launch_bounds__(256) void k_reduce(
    const int* __restrict__ trans, const __bf16* __restrict__ h_buf,
    const float* __restrict__ c_buf, const __bf16* __restrict__ Wrb,
    const float* __restrict__ br, __bf16* __restrict__ hs, float* __restrict__ cs,
    float* __restrict__ out, int nsteps) {
  __shared__ int    s_sp[Bb];
  __shared__ int    s_bp[Bb];
  __shared__ __bf16 s_wr[80][WR_PAD];    // K-first-half of this slice's Wr rows

  int tid = threadIdx.x;
  int lane = tid & 31, wave = tid >> 5;
  int half = lane >> 4, l16 = lane & 15;
  int p  = blockIdx.x;                   // slice id 0..15
  int c0 = p * 16;

  if (tid < Bb) { s_sp[tid] = 0; s_bp[tid] = Tt; }

  // -------------------------------------------------------------------------
  // Stage Wr[n, 0:256] for the 80 rows n = s*Hh + c0 + j (s=0..4, j=0..15).
  // TDM: one 3D-tile descriptor (X=256 elems, Y=16 rows @stride 512,
  // Z=5 gate groups @stride 256*512), hardware-padded so each 256-bf16 row
  // lands at LDS stride 264 bf16 (pad_interval=6 -> every 128 DWORDs,
  // pad_amount=3 -> 4 DWORDs).
  // -------------------------------------------------------------------------
#if HAVE_TDM
  if (wave == 0) {
    unsigned long long ga =
        (unsigned long long)(uintptr_t)(Wrb + (size_t)c0 * (2 * Hh));
    unsigned int lds_off = (unsigned int)(uintptr_t)&s_wr[0][0]; // low 32b = LDS byte offset
    v4u g0 = { 1u,                                   // count=1, user descriptor
               lds_off,                               // lds_addr [63:32]
               (unsigned int)ga,                      // global_addr[31:0]
               ((unsigned int)(ga >> 32) & 0x01FFFFFFu) | (2u << 30) }; // [56:32] + type=2
    v8i g1 = { (int)((1u << 16) |                    // data_size = 2 bytes
                     (1u << 20) |                    // pad_enable
                     (6u << 22) |                    // pad_interval: 128 DWORDs
                     (3u << 25)),                    // pad_amount: 4 DWORDs
               (int)(512u << 16),                    // tensor_dim0[15:0] @ bits 63:48
               (int)(1280u << 16),                   // tensor_dim0 hi=0, tensor_dim1[15:0]
               (int)(256u << 16),                    // tensor_dim1 hi=0, tile_dim0=256
               (int)(16u | (5u << 16)),              // tile_dim1=16, tile_dim2=5
               (int)512,                             // tensor_dim0_stride[31:0]
               0,                                    // stride0 hi=0, dim1_stride[15:0]=0
               (int)2 };                             // tensor_dim1_stride[47:16] = 2 (=131072)
    v4i g2 = { 5, 0, 0, 0 };                         // tensor_dim2=5
    v4i g3 = { 0, 0, 0, 0 };
#if __clang_major__ >= 23
    v8i g4 = { 0, 0, 0, 0, 0, 0, 0, 0 };
    __builtin_amdgcn_tensor_load_to_lds(g0, g1, g2, g3, g4, 0);
#else
    __builtin_amdgcn_tensor_load_to_lds(g0, g1, g2, g3, 0);
#endif
    __builtin_amdgcn_s_wait_tensorcnt(0);
  }
  __syncthreads();
#else
  for (int idx = tid; idx < 80 * 32; idx += 256) {
    int r = idx >> 5, ch = idx & 31;
    int ng = (r >> 4) * Hh + c0 + (r & 15);
    *(bf16x8*)&s_wr[r][ch * 8] = *(const bf16x8*)&Wrb[(size_t)ng * (2 * Hh) + ch * 8];
  }
  __syncthreads();
#endif

  float brv[5];
#pragma unroll
  for (int s = 0; s < 5; ++s) brv[s] = br[s * Hh + c0 + l16];

  int mrow = wave * 16 + l16;            // batch row this lane's A fragment covers
  const __bf16* wr_g[5];                 // global pointers for K second half
#pragma unroll
  for (int s = 0; s < 5; ++s)
    wr_g[s] = Wrb + (size_t)(s * Hh + c0 + l16) * (2 * Hh) + Hh;

  for (int step = 0; step < nsteps; ++step) {
    int spb   = s_sp[mrow];
    int slotL = imax(spb - 2, 0), slotR = imax(spb - 1, 0);
    const __bf16* rowL = hs + (size_t)(slotL * Bb + mrow) * Hh;
    const __bf16* rowR = hs + (size_t)(slotR * Bb + mrow) * Hh;

    v8f zero = {};
    v8f acc[5];
#pragma unroll
    for (int s = 0; s < 5; ++s) acc[s] = zero;

    // K = 512: kt<8 -> A=hl, B from LDS; kt>=8 -> A=hr, B from L2.
    for (int kt = 0; kt < 16; ++kt) {
      int k0 = (kt & 7) * 32;
      v16bf a = ld_a_frag(kt < 8 ? rowL : rowR, k0, half);
      if (kt < 8) {
#pragma unroll
        for (int s = 0; s < 5; ++s) {
          v16bf b = ld_b_frag(&s_wr[s * 16 + l16][0], k0, half);
          acc[s] = wmma_bf16(a, b, acc[s]);
        }
      } else {
#pragma unroll
        for (int s = 0; s < 5; ++s) {
          v16bf b = ld_b_frag(wr_g[s], k0, half);
          acc[s] = wmma_bf16(a, b, acc[s]);
        }
      }
    }

    // All cross-WGP stack reads are consumed; wait for every WG before the
    // in-place stack-top overwrite (WAR hazard across the cluster).
    __syncthreads();
    __builtin_amdgcn_s_cluster_barrier();

#pragma unroll
    for (int i = 0; i < 8; ++i) {
      int b   = wave * 16 + i + 8 * half;
      int col = c0 + l16;
      int tr  = trans[(size_t)step * Bb + b];
      int sb  = s_sp[b], bb = s_bp[b];
      int sL  = imax(sb - 2, 0), sR = imax(sb - 1, 0);
      float pi = acc[0][i] + brv[0];
      float pfl = acc[1][i] + brv[1];
      float pfr = acc[2][i] + brv[2];
      float pg = acc[3][i] + brv[3];
      float po = acc[4][i] + brv[4];
      float cl = cs[(size_t)(sL * Bb + b) * Hh + col];
      float cr = cs[(size_t)(sR * Bb + b) * Hh + col];
      float cn = sigmf(pfl) * cl + sigmf(pfr) * cr + sigmf(pi) * tanhf(pg);
      float hn = sigmf(po) * tanhf(cn);

      int dst; float cw; __bf16 hw;
      if (tr == 0) {                     // shift: push buffers[b, bp-1]
        dst = sb;                        // new_sp-1 = sp
        int bsrc = imax(bb - 1, 0);
        cw = c_buf[(size_t)(b * Tt + bsrc) * Hh + col];
        hw = h_buf[(size_t)(b * Tt + bsrc) * Hh + col];
      } else {                           // reduce: overwrite stack[sp-2]
        dst = imax(sb - 2, 0);
        cw = cn; hw = (__bf16)hn;
      }
      cs[(size_t)(dst * Bb + b) * Hh + col] = cw;
      hs[(size_t)(dst * Bb + b) * Hh + col] = hw;
    }

    // Publish this WG's slice at device scope, update pointers, resync.
    __builtin_amdgcn_fence(__ATOMIC_RELEASE, "agent");
    __syncthreads();
    if (tid < Bb) {
      int tr = trans[(size_t)step * Bb + tid];
      if (tr == 0) { s_sp[tid] += 1; s_bp[tid] -= 1; }
      else         { s_sp[tid] -= 1; }
    }
    __builtin_amdgcn_s_cluster_barrier();
    __builtin_amdgcn_fence(__ATOMIC_ACQUIRE, "agent");
    __syncthreads();
  }

  // Root state: h half of stack top, this WG's column slice.
  for (int idx = tid; idx < Bb * 16; idx += 256) {
    int b = idx >> 4, j = idx & 15;
    int top = imax(s_sp[b] - 1, 0);
    out[(size_t)b * Hh + c0 + j] =
        (float)hs[(size_t)(top * Bb + b) * Hh + c0 + j];
  }
}

// ---------------------------------------------------------------------------
extern "C" void kernel_launch(void* const* d_in, const int* in_sizes, int n_in,
                              void* d_out, int out_size, void* d_ws, size_t ws_size,
                              hipStream_t stream) {
  (void)n_in; (void)out_size; (void)ws_size;
  const float* x     = (const float*)d_in[0];
  const int*   trans = (const int*)d_in[1];
  const float* Wx    = (const float*)d_in[2];
  const float* bx    = (const float*)d_in[3];
  const float* Wg    = (const float*)d_in[4];
  const float* bg    = (const float*)d_in[5];
  const float* Wr    = (const float*)d_in[6];
  const float* br    = (const float*)d_in[7];
  float* out = (float*)d_out;

  char* ws = (char*)d_ws;
  size_t off = 0;
  auto carve = [&](size_t bytes) -> void* {
    void* ptr = ws + off;
    off += (bytes + 255) & ~(size_t)255;
    return ptr;
  };
  __bf16* xb   = (__bf16*)carve((size_t)Bb * Tt * Ee * 2);
  __bf16* Wxb  = (__bf16*)carve((size_t)Hh * Ee * 2);
  __bf16* Wgb  = (__bf16*)carve((size_t)Hh * Ee * 2);
  __bf16* Wrb  = (__bf16*)carve((size_t)5 * Hh * 2 * Hh * 2);
  __bf16* hbuf = (__bf16*)carve((size_t)Bb * Tt * Hh * 2);
  float*  cbuf = (float*) carve((size_t)Bb * Tt * Hh * 4);
  __bf16* hstk = (__bf16*)carve((size_t)Tt * Bb * Hh * 2);
  float*  cstk = (float*) carve((size_t)Tt * Bb * Hh * 4);

  int nsteps = in_sizes[1] / Bb;   // (2T-1)

  // fp32 -> bf16 conversions
  {
    int n;
    n = Bb * Tt * Ee;      k_cvt_bf16<<<(n + 255) / 256, 256, 0, stream>>>(x,  xb,  n);
    n = Hh * Ee;           k_cvt_bf16<<<(n + 255) / 256, 256, 0, stream>>>(Wx, Wxb, n);
    n = Hh * Ee;           k_cvt_bf16<<<(n + 255) / 256, 256, 0, stream>>>(Wg, Wgb, n);
    n = 5 * Hh * 2 * Hh;   k_cvt_bf16<<<(n + 255) / 256, 256, 0, stream>>>(Wr, Wrb, n);
  }
  // zero stack slots 0..1 (reference stack starts at zero; clamped reads hit these)
  {
    int n = 2 * Bb * Hh;
    k_zero_stack<<<(n + 255) / 256, 256, 0, stream>>>(hstk, cstk, n);
  }
  // Phase 1: buffers
  {
    dim3 grid(Bb * Tt / 16 / 8, Hh / 16);   // (256, 16)
    k_buffers<<<grid, 256, 0, stream>>>(xb, Wxb, Wgb, bx, bg, cbuf, hbuf);
  }
  // Phase 2: persistent cluster chain
  k_reduce<<<NSLICE, 256, 0, stream>>>(trans, hbuf, cbuf, Wrb, br,
                                       hstk, cstk, out, nsteps);
}
